// MultiheadAttention_31267361915522
// MI455X (gfx1250) — matrix operand order repaired and checked by reading, not compiled, not run on hardware
//
#include <hip/hip_runtime.h>
#include <hip/hip_bf16.h>

// ---------------------------------------------------------------------------
// MI455X (gfx1250) multihead attention w/ relative-position gating.
// bf16 WMMA (v_wmma_f32_16x16x32_bf16) for all GEMMs + flash attention
// (online softmax) so the 256MB (B,H,T,T) score tensor is never materialized.
// ~52 GFLOP vs ~120MB HBM traffic (~5us @ 23.3TB/s): compute-bound -> matrix
// pipe. Tile staging uses gfx1250 GLOBAL_LOAD_ASYNC_TO_LDS_B128 (ASYNCcnt)
// with double buffering so loads overlap WMMA work; one barrier per K-step.
// Identity used: softmax((aw-rowmax)*ALPHA + gate*pb) == softmax(q0.k*SCALING
// + gate*pb) since the rowmax term is a row-constant shift.
// ---------------------------------------------------------------------------

#define TT   1024
#define BB   4
#define EE   1024
#define HH   16
#define DH   64
#define SCALING 0.125f     // Dh^-0.5

typedef __attribute__((ext_vector_type(16))) __bf16 v16bf;
typedef __attribute__((ext_vector_type(8)))  float  v8f;
typedef int v4i128 __attribute__((vector_size(16)));   // matches builtin param

#define WMMA_BF16(a, b, c) \
  __builtin_amdgcn_wmma_f32_16x16x32_bf16(false, (a), false, (b), (short)0, (c), false, false)

// ---- async global->LDS staging (gfx1250) with sync fallback ---------------
#if defined(__has_builtin)
#if __has_builtin(__builtin_amdgcn_global_load_async_to_lds_b128) && \
    __has_builtin(__builtin_amdgcn_s_wait_asynccnt)
#define USE_ASYNC_LDS 1
#endif
#endif
#ifndef USE_ASYNC_LDS
#define USE_ASYNC_LDS 0
#endif

#if USE_ASYNC_LDS
#define STAGE_B128(gp, lp)                                                  \
  __builtin_amdgcn_global_load_async_to_lds_b128(                           \
      (__attribute__((address_space(1))) v4i128*)(gp),                      \
      (__attribute__((address_space(3))) v4i128*)(lp), 0, 0)
#define WAIT_STAGE() __builtin_amdgcn_s_wait_asynccnt(0)
#else
#define STAGE_B128(gp, lp) (*(uint4*)(lp) = *(const uint4*)(gp))
#define WAIT_STAGE() ((void)0)
#endif

__device__ __forceinline__ unsigned short f2bf(float f) {
  union { float f; unsigned u; } x; x.f = f;
  unsigned r = x.u + 0x7FFFu + ((x.u >> 16) & 1u);  // round-to-nearest-even
  return (unsigned short)(r >> 16);
}
__device__ __forceinline__ float bf2f(unsigned short u) {
  union { unsigned u; float f; } x; x.u = ((unsigned)u) << 16; return x.f;
}

// ---------------------------------------------------------------------------
// Kernel 1: f32 -> bf16 conversion of query (4M elems) and wq/wk/wv/wo (1M ea)
// ---------------------------------------------------------------------------
__global__ void cvt_bf16_kernel(const float* __restrict__ query,
                                const float* __restrict__ wq,
                                const float* __restrict__ wk,
                                const float* __restrict__ wv,
                                const float* __restrict__ wo,
                                unsigned short* __restrict__ dst) {
  size_t idx = (size_t)blockIdx.x * blockDim.x + threadIdx.x;
  const float* src; size_t off;
  if (idx < 4194304u) { src = query; off = idx; }
  else {
    size_t r = idx - 4194304u;
    int w = (int)(r >> 20); off = r & 1048575u;
    src = (w == 0) ? wq : (w == 1) ? wk : (w == 2) ? wv : wo;
  }
  dst[idx] = f2bf(src[off]);
}

// ---------------------------------------------------------------------------
// Kernel 2: relative-position bias table; pb depends only on delta = s - t.
// btab[h][delta+1023], 16 x 2047, stride 2048.
// ---------------------------------------------------------------------------
__global__ void btab_kernel(const float* __restrict__ rel_emb,
                            float* __restrict__ btab) {
  int d = blockIdx.x * blockDim.x + threadIdx.x;
  if (d >= 2047) return;
  int rel = d - 1023;                    // rel = s - t
  int bucket = (rel > 0) ? 160 : 0;      // NUM_BUCKETS/2
  int a = (rel < 0) ? -rel : rel;
  int idx;
  if (a < 80) idx = a;                   // max_exact
  else {
    idx = 80 + (int)(__logf((float)a * (1.0f / 80.0f)) * (80.0f / __logf(10.0f)));
    if (idx > 159) idx = 159;
  }
  bucket += idx;
  #pragma unroll
  for (int h = 0; h < HH; ++h)
    btab[h * 2048 + d] = rel_emb[bucket * HH + h];
}

// ---------------------------------------------------------------------------
// Kernel 3: collapse grep_w (8x64) into two 64-vectors (reshape(2,4).sum is
// linear): gvec[0..63], gvec[64..127]; summed biases at [128],[129].
// ---------------------------------------------------------------------------
__global__ void gvec_kernel(const float* __restrict__ grep_w,
                            const float* __restrict__ grep_b,
                            float* __restrict__ gvec) {
  int d = threadIdx.x;  // 64 threads
  float s0 = 0.f, s1 = 0.f;
  #pragma unroll
  for (int j = 0; j < 4; ++j) { s0 += grep_w[j * DH + d]; s1 += grep_w[(4 + j) * DH + d]; }
  gvec[d] = s0; gvec[DH + d] = s1;
  if (d == 0) {
    float b0 = 0.f, b1 = 0.f;
    #pragma unroll
    for (int j = 0; j < 4; ++j) { b0 += grep_b[j]; b1 += grep_b[4 + j]; }
    gvec[128] = b0; gvec[129] = b1;
  }
}

// ---------------------------------------------------------------------------
// GEMM core: per block C(128x64) = X(4096x1024) @ W^T, W row-major (n,k).
// 8 waves x (16 rows x 64 cols); K-step 32; double-buffered async LDS staging.
// WMMA operand layouts used here (contiguous k, same order in A and B):
//   A 16x32 bf16 : lane row = lane&15, K = (lane<16)?0..15:16..31
//                  -> two contiguous 16B loads.
//   B 32x16 bf16 : lane col = lane&15, K = (lane<16)?0..15:16..31 -> 32B.
//   C/D 16x16 f32: lane col = lane&15, rows = ((lane<16)?0:8)+vgpr.
// ---------------------------------------------------------------------------
__device__ __forceinline__ void gemm_core(const unsigned short* __restrict__ X,
                                          const unsigned short* __restrict__ W,
                                          unsigned short* Xs,   // 2 x 128*32
                                          unsigned short* Wsh,  // 2 x 64*32
                                          int m0, int n0, int tid, v8f acc[4]) {
  const int wave = tid >> 5, lane = tid & 31;
  const int ar  = wave * 16 + (lane & 15);
  const int kb8 = (lane < 16) ? 0 : 8;
  const int bn  = lane & 15;
  const int bkb = (lane < 16) ? 0 : 16;

  auto stage = [&](int k0, int buf) {
    unsigned short* xd = Xs  + buf * (128 * 32);
    unsigned short* wd = Wsh + buf * (64 * 32);
    #pragma unroll
    for (int i = 0; i < 2; ++i) {               // X block: 128x32 bf16 = 8KB
      int c = tid + 256 * i;
      int r = c >> 2, co = (c & 3) * 8;
      STAGE_B128(&X[(size_t)(m0 + r) * EE + k0 + co], &xd[r * 32 + co]);
    }
    {                                           // W block: 64x32 bf16 = 4KB
      int r = tid >> 2, co = (tid & 3) * 8;
      STAGE_B128(&W[(size_t)(n0 + r) * EE + k0 + co], &wd[r * 32 + co]);
    }
  };

  stage(0, 0);
  for (int k0 = 0; k0 < EE; k0 += 32) {
    const int buf = (k0 >> 5) & 1;
    WAIT_STAGE();            // my async writes for buf landed in LDS
    __syncthreads();         // everyone's landed; prior readers of buf^1 done
    if (k0 + 32 < EE) stage(k0 + 32, buf ^ 1);   // overlap with compute
    const unsigned short* xs = Xs  + buf * (128 * 32);
    const unsigned short* ws = Wsh + buf * (64 * 32);
    v16bf a;
    ((uint4*)&a)[0] = *(const uint4*)(&xs[ar * 32 + bkb]);
    ((uint4*)&a)[1] = *(const uint4*)(&xs[ar * 32 + bkb + 8]);
    #pragma unroll
    for (int dt = 0; dt < 4; ++dt) {
      v16bf bm;
      ((uint4*)&bm)[0] = *(const uint4*)(&ws[(dt * 16 + bn) * 32 + bkb]);
      ((uint4*)&bm)[1] = *(const uint4*)(&ws[(dt * 16 + bn) * 32 + bkb + 8]);
      acc[dt] = WMMA_BF16(a, bm, acc[dt]);
    }
  }
}

// Projection GEMM: blockIdx.z selects {Q,K,V}. Epilogue scatters:
//  mode 0: Qbf[b,h,t,d] = bf16((x@wq^T + bq) * SCALING)
//  mode 1: Kbf[b,h,t,d]
//  mode 2: Vtb[b,h,d,t]  (transposed so PV B-tiles are contiguous)
__global__ __launch_bounds__(256) void gemm_proj_kernel(
    const unsigned short* __restrict__ X,
    const unsigned short* __restrict__ Wq, const unsigned short* __restrict__ Wk,
    const unsigned short* __restrict__ Wv,
    const float* __restrict__ bq, const float* __restrict__ bk,
    const float* __restrict__ bv,
    unsigned short* __restrict__ Qo, unsigned short* __restrict__ Ko,
    unsigned short* __restrict__ Vo) {
  __shared__ __align__(16) unsigned short Xs[2 * 128 * 32];
  __shared__ __align__(16) unsigned short Wsh[2 * 64 * 32];
  const int tid = threadIdx.x;
  const int mode = blockIdx.z;
  const int n0 = blockIdx.x * 64;
  const int m0 = blockIdx.y * 128;
  const unsigned short* W = (mode == 0) ? Wq : (mode == 1) ? Wk : Wv;
  const float* bias = (mode == 0) ? bq : (mode == 1) ? bk : bv;
  v8f acc[4] = {};
  gemm_core(X, W, Xs, Wsh, m0, n0, tid, acc);
  const int wave = tid >> 5, lane = tid & 31;
  const int nloc = lane & 15, mbase = (lane < 16) ? 0 : 8;
  #pragma unroll
  for (int dt = 0; dt < 4; ++dt) {
    int e = n0 + dt * 16 + nloc;
    float bval = bias[e];
    int h = e >> 6, d = e & 63;
    #pragma unroll
    for (int v = 0; v < 8; ++v) {
      int r = m0 + wave * 16 + mbase + v;     // r = t*B + b
      int t = r >> 2, b = r & 3;
      float val = acc[dt][v] + bval;
      size_t bh = (size_t)(b * HH + h);
      if (mode == 0)      Qo[(bh * TT + t) * DH + d] = f2bf(val * SCALING);
      else if (mode == 1) Ko[(bh * TT + t) * DH + d] = f2bf(val);
      else                Vo[(bh * DH + d) * TT + t] = f2bf(val);
    }
  }
}

// Final projection: out = attn @ wo^T + bo, f32 output (T,B,E), row = t*B+b.
__global__ __launch_bounds__(256) void gemm_out_kernel(
    const unsigned short* __restrict__ X, const unsigned short* __restrict__ W,
    const float* __restrict__ bo, float* __restrict__ out) {
  __shared__ __align__(16) unsigned short Xs[2 * 128 * 32];
  __shared__ __align__(16) unsigned short Wsh[2 * 64 * 32];
  const int tid = threadIdx.x;
  const int n0 = blockIdx.x * 64;
  const int m0 = blockIdx.y * 128;
  v8f acc[4] = {};
  gemm_core(X, W, Xs, Wsh, m0, n0, tid, acc);
  const int wave = tid >> 5, lane = tid & 31;
  const int nloc = lane & 15, mbase = (lane < 16) ? 0 : 8;
  #pragma unroll
  for (int dt = 0; dt < 4; ++dt) {
    int e = n0 + dt * 16 + nloc;
    float bval = bo[e];
    #pragma unroll
    for (int v = 0; v < 8; ++v) {
      int r = m0 + wave * 16 + mbase + v;
      out[(size_t)r * EE + e] = acc[dt][v] + bval;
    }
  }
}

// ---------------------------------------------------------------------------
// Kernel 5: gates. gate[b,h,t] = ga*(gb*grep_a[h]-1)+2, ga/gb =
// sigmoid(q0 . gvec + gbias). Qbf holds q0*SCALING -> multiply by 8.
// ---------------------------------------------------------------------------
__global__ void gate_kernel(const unsigned short* __restrict__ Qbf,
                            const float* __restrict__ gvec,
                            const float* __restrict__ grep_a,
                            float* __restrict__ gate) {
  int idx = blockIdx.x * blockDim.x + threadIdx.x;   // B*H*T = 65536
  int h = (idx >> 10) & 15;
  const unsigned short* q = Qbf + (size_t)idx * DH;
  float la = gvec[128], lb = gvec[129];
  #pragma unroll 8
  for (int d = 0; d < DH; ++d) {
    float qv = bf2f(q[d]) * 8.0f;                    // undo SCALING
    la += qv * gvec[d];
    lb += qv * gvec[DH + d];
  }
  float ga = 1.0f / (1.0f + __expf(-la));
  float gb = 1.0f / (1.0f + __expf(-lb));
  gate[idx] = ga * (gb * grep_a[h] - 1.0f) + 2.0f;
}

// ---------------------------------------------------------------------------
// Kernel 6: flash attention. Grid (T/128, B*H); 256 threads = 8 waves, each
// wave owns 16 query rows and the full Dh=64 output (4 C tiles). Per s-block
// of 32 keys: 4 WMMA scores, fused gate*bias, online softmax (16-lane
// half-wave shuffle reductions), P bf16 via per-wave LDS C->A transpose,
// 4 WMMA PV. K/V^T staging is double-buffered async-to-LDS.
// ---------------------------------------------------------------------------
__global__ __launch_bounds__(256) void flash_attn_kernel(
    const unsigned short* __restrict__ Qbf,   // [B*H][T][64]  (q0*SCALING)
    const unsigned short* __restrict__ Kbf,   // [B*H][T][64]
    const unsigned short* __restrict__ Vtb,   // [B*H][64][T]
    const float* __restrict__ gate,           // [B*H][T]
    const float* __restrict__ btab,           // [H][2048]
    unsigned short* __restrict__ attnb) {     // [4096][1024] row=t*B+b
  __shared__ __align__(16) unsigned short Ks[2][32 * 64];
  __shared__ __align__(16) unsigned short Vs[2][64 * 32];
  __shared__ __align__(16) unsigned short Ps[8][16 * 32];
  __shared__ __align__(16) float Bt[2048];
  const int tid = threadIdx.x, wave = tid >> 5, lane = tid & 31;
  const int bh = blockIdx.y;
  const int h = bh & 15;
  const int t0 = blockIdx.x * 128;

  const unsigned short* Qb = Qbf + (size_t)bh * TT * DH;
  const unsigned short* Kb = Kbf + (size_t)bh * TT * DH;
  const unsigned short* Vb = Vtb + (size_t)bh * DH * TT;

  // stage the head's bias row (2048 floats = 512 x 16B; last elem unused)
  #pragma unroll
  for (int i = 0; i < 2; ++i) {
    int c = tid + 256 * i;
    STAGE_B128(&btab[h * 2048 + c * 4], &Bt[c * 4]);
  }

  const int nloc  = lane & 15;
  const int kb8   = (lane < 16) ? 0 : 8;     // unused; A loads also use bkb16
  const int bkb16 = (lane < 16) ? 0 : 16;    // B-layout K base
  const int mbase = (lane < 16) ? 0 : 8;     // C-layout row base

  // Q tile in A-layout registers (2 K-steps over Dh=64)
  const int arow = t0 + wave * 16 + nloc;
  v16bf qa0, qa1;
  ((uint4*)&qa0)[0] = *(const uint4*)(&Qb[(size_t)arow * DH + bkb16]);
  ((uint4*)&qa0)[1] = *(const uint4*)(&Qb[(size_t)arow * DH + bkb16 + 8]);
  ((uint4*)&qa1)[0] = *(const uint4*)(&Qb[(size_t)arow * DH + 32 + bkb16]);
  ((uint4*)&qa1)[1] = *(const uint4*)(&Qb[(size_t)arow * DH + 32 + bkb16 + 8]);

  auto stageKV = [&](int s0, int buf) {
    int r = tid >> 3, co = (tid & 7) * 8;       // K: 32x64 bf16 = 4KB
    STAGE_B128(&Kb[(size_t)(s0 + r) * DH + co], &Ks[buf][r * 64 + co]);
    int r2 = tid >> 2, co2 = (tid & 3) * 8;     // V^T: 64x32 bf16 = 4KB
    STAGE_B128(&Vb[(size_t)r2 * TT + s0 + co2], &Vs[buf][r2 * 32 + co2]);
  };

  float g8[8], m[8], l[8];
  v8f o[4] = {};
  #pragma unroll
  for (int v = 0; v < 8; ++v) {
    g8[v] = gate[(size_t)bh * TT + t0 + wave * 16 + mbase + v];
    m[v] = -1e30f; l[v] = 0.f;
  }

  stageKV(0, 0);
  for (int s0 = 0; s0 < TT; s0 += 32) {
    const int buf = (s0 >> 5) & 1;
    WAIT_STAGE();
    __syncthreads();
    if (s0 + 32 < TT) stageKV(s0 + 32, buf ^ 1);   // overlap with compute
    const unsigned short* ks = Ks[buf];
    const unsigned short* vs = Vs[buf];

    // scores: two 16-col subtiles, each K=64 (2 WMMA each)
    v8f c0 = {}, c1 = {};
    {
      v16bf bm;
      ((uint4*)&bm)[0] = *(const uint4*)(&ks[nloc * 64 + bkb16]);
      ((uint4*)&bm)[1] = *(const uint4*)(&ks[nloc * 64 + bkb16 + 8]);
      c0 = WMMA_BF16(qa0, bm, c0);
      ((uint4*)&bm)[0] = *(const uint4*)(&ks[nloc * 64 + 32 + bkb16]);
      ((uint4*)&bm)[1] = *(const uint4*)(&ks[nloc * 64 + 32 + bkb16 + 8]);
      c0 = WMMA_BF16(qa1, bm, c0);
      ((uint4*)&bm)[0] = *(const uint4*)(&ks[(16 + nloc) * 64 + bkb16]);
      ((uint4*)&bm)[1] = *(const uint4*)(&ks[(16 + nloc) * 64 + bkb16 + 8]);
      c1 = WMMA_BF16(qa0, bm, c1);
      ((uint4*)&bm)[0] = *(const uint4*)(&ks[(16 + nloc) * 64 + 32 + bkb16]);
      ((uint4*)&bm)[1] = *(const uint4*)(&ks[(16 + nloc) * 64 + 32 + bkb16 + 8]);
      c1 = WMMA_BF16(qa1, bm, c1);
    }

    // fused gate * relative-position bias + online softmax update
    unsigned short* P = &Ps[wave][0];
    #pragma unroll
    for (int v = 0; v < 8; ++v) {
      int t = t0 + wave * 16 + mbase + v;
      int s = s0 + nloc;
      c0[v] += g8[v] * Bt[s - t + 1023];
      c1[v] += g8[v] * Bt[s + 16 - t + 1023];
      float x = fmaxf(c0[v], c1[v]);
      #pragma unroll
      for (int off = 1; off < 16; off <<= 1) x = fmaxf(x, __shfl_xor(x, off, 32));
      float mn = fmaxf(m[v], x);
      float sc = __expf(m[v] - mn);
      m[v] = mn;
      float p0 = __expf(c0[v] - mn);
      float p1 = __expf(c1[v] - mn);
      float ps = p0 + p1;
      #pragma unroll
      for (int off = 1; off < 16; off <<= 1) ps += __shfl_xor(ps, off, 32);
      l[v] = l[v] * sc + ps;
      o[0][v] *= sc; o[1][v] *= sc; o[2][v] *= sc; o[3][v] *= sc;
      int r = mbase + v;
      P[r * 32 + nloc]      = f2bf(p0);   // C-layout -> row-major LDS
      P[r * 32 + 16 + nloc] = f2bf(p1);
    }
    __syncthreads();   // P coherent before PV (Vs[buf] still valid)

    // P as A-matrix (16x32), V^T tiles as B, accumulate 4 output tiles
    v16bf pa;
    ((uint4*)&pa)[0] = *(const uint4*)(&P[nloc * 32 + bkb16]);
    ((uint4*)&pa)[1] = *(const uint4*)(&P[nloc * 32 + bkb16 + 8]);
    #pragma unroll
    for (int dt = 0; dt < 4; ++dt) {
      v16bf bm;
      ((uint4*)&bm)[0] = *(const uint4*)(&vs[(dt * 16 + nloc) * 32 + bkb16]);
      ((uint4*)&bm)[1] = *(const uint4*)(&vs[(dt * 16 + nloc) * 32 + bkb16 + 8]);
      o[dt] = WMMA_BF16(pa, bm, o[dt]);
    }
  }

  // normalize and scatter to attn buffer (row = t*B+b, col = h*64+d), bf16
  const int b = bh >> 4;
  #pragma unroll
  for (int v = 0; v < 8; ++v) {
    float inv = 1.0f / l[v];
    int t = t0 + wave * 16 + mbase + v;
    unsigned short* orow = attnb + ((size_t)t * BB + b) * EE + h * DH;
    #pragma unroll
    for (int dt = 0; dt < 4; ++dt)
      orow[dt * 16 + nloc] = f2bf(o[dt][v] * inv);
  }
}

// ---------------------------------------------------------------------------
// Launch. Workspace layout:
//   [0,16M)   : Xbf(8M) + Wq/Wk/Wv/Wo bf16 (2M each); Xbf reused as attn buf.
//   [16M,40M) : Qbf / Kbf / Vtb (8M each)
//   [40M,..)  : gate (256K), btab (128K), gvec (1K)
// ---------------------------------------------------------------------------
extern "C" void kernel_launch(void* const* d_in, const int* in_sizes, int n_in,
                              void* d_out, int out_size, void* d_ws, size_t ws_size,
                              hipStream_t stream) {
  const float* query  = (const float*)d_in[0];
  const float* wq     = (const float*)d_in[1];
  const float* bq     = (const float*)d_in[2];
  const float* wk     = (const float*)d_in[3];
  const float* bk     = (const float*)d_in[4];
  const float* wv     = (const float*)d_in[5];
  const float* bv     = (const float*)d_in[6];
  const float* wo     = (const float*)d_in[7];
  const float* bo     = (const float*)d_in[8];
  const float* rel_emb= (const float*)d_in[9];
  const float* grep_w = (const float*)d_in[10];
  const float* grep_b = (const float*)d_in[11];
  const float* grep_a = (const float*)d_in[12];

  char* ws = (char*)d_ws;
  unsigned short* Xbf = (unsigned short*)ws;        // 4M elems
  unsigned short* Wqb = Xbf + 4194304;
  unsigned short* Wkb = Xbf + 5242880;
  unsigned short* Wvb = Xbf + 6291456;
  unsigned short* Wob = Xbf + 7340032;
  unsigned short* Qbf = Xbf + 8388608;              // at 16MB
  unsigned short* Kbf = Qbf + 4194304;
  unsigned short* Vtb = Kbf + 4194304;
  unsigned short* Att = Xbf;                        // alias: Xbf dead post-proj
  float* gatep = (float*)(ws + (size_t)40 * 1048576);
  float* btab  = gatep + 65536;
  float* gvec  = btab + 16 * 2048;

  cvt_bf16_kernel<<<32768, 256, 0, stream>>>(query, wq, wk, wv, wo, Xbf);
  btab_kernel<<<8, 256, 0, stream>>>(rel_emb, btab);
  gvec_kernel<<<1, 64, 0, stream>>>(grep_w, grep_b, gvec);

  dim3 gp(16, 32, 3);     // Nblk=1024/64, Mblk=4096/128, {Q,K,V}
  gemm_proj_kernel<<<gp, 256, 0, stream>>>(Xbf, Wqb, Wkb, Wvb, bq, bk, bv,
                                           Qbf, Kbf, Vtb);
  gate_kernel<<<256, 256, 0, stream>>>(Qbf, gvec, grep_a, gatep);

  dim3 gf(8, 64);         // T/128 row-blocks x (B*H)
  flash_attn_kernel<<<gf, 256, 0, stream>>>(Qbf, Kbf, Vtb, gatep, btab, Att);

  dim3 go(16, 32);
  gemm_out_kernel<<<go, 256, 0, stream>>>(Att, Wob, bo, (float*)d_out);
}
